// MultiHeadAttentionLayer_68865505624542
// MI455X (gfx1250) — matrix operand order repaired and checked by reading, not compiled, not run on hardware
//
#include <hip/hip_runtime.h>

#define EMBED 1024
#define HEADS 16
#define HDIM  64
#define BATCH 4
#define SEQ   2048

typedef __attribute__((ext_vector_type(16))) __bf16 v16bf;
typedef __attribute__((ext_vector_type(8)))  float  v8f;
typedef __attribute__((ext_vector_type(4)))  unsigned int u32x4;
typedef unsigned short u16;
typedef unsigned int   u32;

struct Frag16  { u16 h[16]; };          // manual element-wise build
struct Frag16V { u32x4 lo, hi; };       // vector-load build (32 bytes)
struct H8      { u16 h[8]; };           // 16-byte packed store

__device__ __forceinline__ u16 f2bf_bits(float f) {
  return __builtin_bit_cast(u16, (__bf16)f);   // native cvt on gfx1250
}
__device__ __forceinline__ float bf2f(u16 h) {
  u32 u = ((u32)h) << 16;
  return __builtin_bit_cast(float, u);
}
__device__ __forceinline__ v16bf frag_cast(const Frag16& f) {
  return __builtin_bit_cast(v16bf, f);
}
// B-fragment / generic: 16 contiguous bf16 at p (16B-aligned)
__device__ __forceinline__ v16bf frag_b(const u16* p) {
  Frag16V f{*(const u32x4*)p, *(const u32x4*)(p + 8)};
  return __builtin_bit_cast(v16bf, f);
}
// A-fragment: 8 contiguous bf16 at p and 8 at p+16 (each 16B-aligned)
__device__ __forceinline__ v16bf frag_a(const u16* p) {
  Frag16V f{*(const u32x4*)p, *(const u32x4*)(p + 16)};
  return __builtin_bit_cast(v16bf, f);
}
__device__ __forceinline__ v8f wmma_bf16(v16bf a, v16bf b, v8f c) {
  return __builtin_amdgcn_wmma_f32_16x16x32_bf16(false, a, false, b, (short)0, c,
                                                 false, false);
}

// ---------------------------------------------------------------------------
// GEMM: C[M,N] = A[M,K] @ W[K,N] + bias.  M=8192, N=K=1024.
// Block = 256 threads (8 waves) -> 128(M) x 64(N) tile.
// W chunk staged TRANSPOSED in LDS ([n][k], 80B pitch) so B-fragments are
// 2x ds_load_b128 per WMMA.
// OUT_MODE: 0 = bf16 [B,H,S,D] (q,k)   1 = bf16 [B,H,D,S] (v, transposed)
//           2 = fp32 [M,N] (final output)
// ---------------------------------------------------------------------------
template <bool A_BF16, int OUT_MODE>
__global__ __launch_bounds__(256)
void gemm_bf16_wmma(const void* __restrict__ Ap, const float* __restrict__ W,
                    const float* __restrict__ bias, void* __restrict__ outp) {
  __shared__ __align__(16) u16 Wlds[64][40];   // [n][k], pitch 80B (16B mult.)

  const int tid  = threadIdx.x;
  const int wave = tid >> 5;
  const int lane = tid & 31;
  const int colbase = lane & 15;
  const int hi      = lane >> 4;

  const int Mbase = blockIdx.x * 128 + wave * 16;
  const int Nbase = blockIdx.y * 64;

  v8f acc[4];
  for (int nt = 0; nt < 4; ++nt) {
    float bv = bias[Nbase + nt * 16 + colbase];
    for (int r = 0; r < 8; ++r) acc[nt][r] = bv;
  }

  const int    arow = Mbase + colbase;         // this lane's A row
  const float* Af = (const float*)Ap;
  const u16*   Ab = (const u16*)Ap;

  for (int kb = 0; kb < EMBED; kb += 32) {
    __syncthreads();   // protect previous iteration's LDS reads
    // W[kb..kb+31][Nbase..Nbase+63] -> LDS transposed, packed u32 stores
    for (int i = 0; i < 4; ++i) {
      int idx = tid + i * 256;                // 0..1023
      int n   = idx & 63;
      int k   = (idx >> 6) * 2;               // even k
      float w0 = W[(size_t)(kb + k)     * EMBED + Nbase + n];
      float w1 = W[(size_t)(kb + k + 1) * EMBED + Nbase + n];
      u32 pk = (u32)f2bf_bits(w0) | ((u32)f2bf_bits(w1) << 16);
      *(u32*)&Wlds[n][k] = pk;
    }
    __syncthreads();

    // A fragment: lanes 0-15 hold K={0..7,16..23}, lanes 16-31 K={8..15,24..31}
    v16bf a;
    const size_t abase = (size_t)arow * EMBED + kb + hi * 8;
    if (A_BF16) {
      a = frag_a(Ab + abase);
      if (kb + 32 < EMBED) __builtin_prefetch(Ab + abase + 32, 0, 0);
    } else {
      Frag16 fa;
      for (int j = 0; j < 8; ++j) fa.h[j]     = f2bf_bits(Af[abase + j]);
      for (int j = 0; j < 8; ++j) fa.h[8 + j] = f2bf_bits(Af[abase + 16 + j]);
      a = frag_cast(fa);
      if (kb + 32 < EMBED) __builtin_prefetch(Af + abase + 32, 0, 0);
    }

    // B fragments: lane holds column n, K=kstart..kstart+7 and kstart+16..kstart+23
    const int kstart = hi * 8;
    for (int nt = 0; nt < 4; ++nt)
      acc[nt] = wmma_bf16(a, frag_a(&Wlds[nt * 16 + colbase][kstart]), acc[nt]);
  }

  // C layout: element r -> row (r + 8*hi), col colbase
  if (OUT_MODE == 0) {
    u16* out = (u16*)outp;
    for (int nt = 0; nt < 4; ++nt) {
      int n = Nbase + nt * 16 + colbase;
      int hh = n >> 6, d = n & 63;
      for (int r = 0; r < 8; ++r) {
        int m = Mbase + r + 8 * hi;
        int b = m >> 11, s = m & (SEQ - 1);
        out[(((size_t)(b * HEADS + hh) * SEQ + s) * HDIM) + d] =
            f2bf_bits(acc[nt][r]);
      }
    }
  } else if (OUT_MODE == 1) {
    u16* out = (u16*)outp;
    int m0 = Mbase + 8 * hi;                   // 8 consecutive rows -> s run
    int b = m0 >> 11, s0 = m0 & (SEQ - 1);     // s0 multiple of 8 -> 16B align
    for (int nt = 0; nt < 4; ++nt) {
      int n = Nbase + nt * 16 + colbase;
      int hh = n >> 6, d = n & 63;
      H8 t;
      for (int r = 0; r < 8; ++r) t.h[r] = f2bf_bits(acc[nt][r]);
      *(u32x4*)&out[((size_t)(b * HEADS + hh) * HDIM + d) * SEQ + s0] =
          __builtin_bit_cast(u32x4, t);
    }
  } else {
    float* out = (float*)outp;
    for (int nt = 0; nt < 4; ++nt) {
      int n = Nbase + nt * 16 + colbase;
      for (int r = 0; r < 8; ++r) {
        int m = Mbase + r + 8 * hi;
        out[(size_t)m * EMBED + n] = acc[nt][r];
      }
    }
  }
}

// ---------------------------------------------------------------------------
// Causal flash attention, one wave per (b, h, 16-row q tile), D = 64.
// 32 keys / iteration: 4 score WMMAs + 1 row-sum WMMA + 4 PV WMMAs.
// Row sums of P are computed as P @ ones(32x16) on the matrix pipe (every
// column of the result equals the row sum, already in accumulator layout),
// replacing one 4-step shuffle butterfly per row.
// kh is [B,H,S,D] (K-fragments = contiguous d-slices), vh is [B,H,D,S]
// (V-fragments = contiguous key-runs) -> all fragments are b128 loads.
// All 8 waves in a block share the same q-tile index -> uniform trip count,
// so __syncthreads() inside the loop is safe (used only for the P bounce).
// ---------------------------------------------------------------------------
__global__ __launch_bounds__(256)
void attn_wmma(const u16* __restrict__ qh, const u16* __restrict__ kh,
               const u16* __restrict__ vh, u16* __restrict__ attn) {
  __shared__ __align__(16) u16 Pbuf[8][16 * 32];   // per-wave 16x32 bf16 P

  const int tid  = threadIdx.x;
  const int wave = tid >> 5;
  const int lane = tid & 31;
  const int colbase = lane & 15;
  const int hi      = lane >> 4;

  const int t  = blockIdx.x;                 // q tile index (uniform in block)
  const int bh = blockIdx.y * 8 + wave;      // 0..B*H-1
  const int qbase = t * 16;
  const size_t kbase = (size_t)bh * SEQ * HDIM;     // kh/qh base
  const size_t vbase = (size_t)bh * HDIM * SEQ;     // vh base (transposed)

  // all-ones bf16 B-fragment for row-sum WMMA
  v16bf ones;
  {
    u32 o = 0x3F803F80u;                     // two bf16 1.0
    u32x4 q4 = {o, o, o, o};
    Frag16V f{q4, q4};
    ones = __builtin_bit_cast(v16bf, f);
  }

  // q A-fragments for d-chunks [0,32) and [32,64), pre-scaled by 1/sqrt(D)
  v16bf qa[2];
  {
    const size_t qrow = kbase + (size_t)(qbase + colbase) * HDIM;
    for (int c = 0; c < 2; ++c) {
      Frag16 f;
      int k0 = c * 32 + hi * 8;
      for (int j = 0; j < 8; ++j)
        f.h[j] = f2bf_bits(0.125f * bf2f(qh[qrow + k0 + j]));
      for (int j = 0; j < 8; ++j)
        f.h[8 + j] = f2bf_bits(0.125f * bf2f(qh[qrow + k0 + 16 + j]));
      qa[c] = frag_cast(f);
    }
  }

  v8f O[4];
  for (int nt = 0; nt < 4; ++nt)
    for (int r = 0; r < 8; ++r) O[nt][r] = 0.0f;
  float mrun[8], lrun[8];
  for (int r = 0; r < 8; ++r) { mrun[r] = -1e30f; lrun[r] = 0.0f; }

  const int nblk = (qbase + 16 + 31) >> 5;   // causal: keys <= qbase+15
  for (int jb = 0; jb < nblk; ++jb) {
    const int c0 = jb * 32;

    // ---- scores: two 16x16 tiles (keys c0..+15, c0+16..+31) ----
    v8f s0, s1;
    for (int r = 0; r < 8; ++r) { s0[r] = 0.0f; s1[r] = 0.0f; }
    for (int tile = 0; tile < 2; ++tile) {
      const u16* krow = kh + kbase + (size_t)(c0 + tile * 16 + colbase) * HDIM;
      for (int c = 0; c < 2; ++c) {
        v16bf kb = frag_a(krow + c * 32 + hi * 8);   // one key's d-slice
        if (tile == 0) s0 = wmma_bf16(qa[c], kb, s0);
        else           s1 = wmma_bf16(qa[c], kb, s1);
      }
    }

    // ---- causal mask (only near the diagonal) ----
    if (c0 + 31 > qbase) {
      for (int r = 0; r < 8; ++r) {
        int row = qbase + r + 8 * hi;
        int k0  = c0 + colbase;
        if (k0 > row)      s0[r] = -1e30f;
        if (k0 + 16 > row) s1[r] = -1e30f;
      }
    }

    // ---- online softmax: max via 16-lane butterfly, exp, rescale ----
    float pr0[8], pr1[8], sc[8];
    for (int r = 0; r < 8; ++r) {
      float mx = fmaxf(s0[r], s1[r]);
      for (int d = 1; d < 16; d <<= 1) mx = fmaxf(mx, __shfl_xor(mx, d, 32));
      float mnew = fmaxf(mrun[r], mx);
      sc[r] = __expf(mrun[r] - mnew);
      pr0[r] = __expf(s0[r] - mnew);
      pr1[r] = __expf(s1[r] - mnew);
      mrun[r] = mnew;
    }
    for (int nt = 0; nt < 4; ++nt)
      for (int r = 0; r < 8; ++r) O[nt][r] *= sc[r];

    // ---- P: C-layout -> A-fragment layout via per-wave LDS bounce ----
    __syncthreads();
    for (int r = 0; r < 8; ++r) {
      int row = r + 8 * hi;
      Pbuf[wave][row * 32 + colbase]      = f2bf_bits(pr0[r]);
      Pbuf[wave][row * 32 + 16 + colbase] = f2bf_bits(pr1[r]);
    }
    __syncthreads();
    v16bf pa = frag_a(&Pbuf[wave][colbase * 32 + hi * 8]);

    // ---- row sums on the matrix pipe: ls = P @ ones ----
    v8f ls;
    for (int r = 0; r < 8; ++r) ls[r] = 0.0f;
    ls = wmma_bf16(pa, ones, ls);
    for (int r = 0; r < 8; ++r) lrun[r] = lrun[r] * sc[r] + ls[r];

    // ---- O += P @ V (4 N-tiles of 16); V^T -> contiguous key runs ----
    for (int nt = 0; nt < 4; ++nt) {
      const u16* vp =
          vh + vbase + (size_t)(nt * 16 + colbase) * SEQ + c0 + hi * 8;
      O[nt] = wmma_bf16(pa, frag_a(vp), O[nt]);
    }
  }

  // ---- epilogue: divide by row sum, write merged-head bf16 [B,S,E] ----
  const int b = bh / HEADS, h = bh % HEADS;
  for (int r = 0; r < 8; ++r) {
    float inv = 1.0f / lrun[r];
    int s = qbase + r + 8 * hi;
    for (int nt = 0; nt < 4; ++nt) {
      int e = h * HDIM + nt * 16 + colbase;
      attn[((size_t)b * SEQ + s) * EMBED + e] = f2bf_bits(O[nt][r] * inv);
    }
  }
}

// ---------------------------------------------------------------------------
extern "C" void kernel_launch(void* const* d_in, const int* in_sizes, int n_in,
                              void* d_out, int out_size, void* d_ws,
                              size_t ws_size, hipStream_t stream) {
  (void)in_sizes; (void)n_in; (void)out_size; (void)ws_size;
  const float* Q  = (const float*)d_in[0];
  const float* K  = (const float*)d_in[1];
  const float* V  = (const float*)d_in[2];
  const float* Wq = (const float*)d_in[3];
  const float* bq = (const float*)d_in[4];
  const float* Wk = (const float*)d_in[5];
  const float* bk = (const float*)d_in[6];
  const float* Wv = (const float*)d_in[7];
  const float* bv = (const float*)d_in[8];
  const float* Wo = (const float*)d_in[9];
  const float* bo = (const float*)d_in[10];
  float* out = (float*)d_out;

  const size_t N = (size_t)BATCH * SEQ * EMBED;   // 8,388,608 elements
  u16* qh = (u16*)d_ws;        // bf16 [B,H,S,D]
  u16* kh = qh + N;            // bf16 [B,H,S,D]
  u16* vh = kh + N;            // bf16 [B,H,D,S]  (transposed)
  u16* at = vh + N;            // bf16 [B,S,E]

  dim3 gg(64, 16), bb(256, 1, 1);
  gemm_bf16_wmma<false, 0><<<gg, bb, 0, stream>>>((const void*)Q, Wq, bq, (void*)qh);
  gemm_bf16_wmma<false, 0><<<gg, bb, 0, stream>>>((const void*)K, Wk, bk, (void*)kh);
  gemm_bf16_wmma<false, 1><<<gg, bb, 0, stream>>>((const void*)V, Wv, bv, (void*)vh);

  attn_wmma<<<dim3(SEQ / 16, (BATCH * HEADS) / 8), bb, 0, stream>>>(qh, kh, vh, at);

  gemm_bf16_wmma<true, 2><<<gg, bb, 0, stream>>>((const void*)at, Wo, bo, (void*)out);
}
